// SelfAttention_20504173871393
// MI455X (gfx1250) — matrix operand order repaired and checked by reading, not compiled, not run on hardware
//
#include <hip/hip_runtime.h>
#include <hip/hip_bf16.h>

// ---------------------------------------------------------------------------
// Self-attention fwd on MI455X (gfx1250, wave32).
// - All GEMM math: v_wmma_f32_16x16x32_bf16 (bf16 in, fp32 accumulate).
// - GEMM tiles staged to LDS with GLOBAL_LOAD_ASYNC_TO_LDS_B128 (ASYNCcnt),
//   enabled when the builtin exists (fallback: synchronous u32 staging).
// - Weights pre-packed into the WMMA B-fragment layout [kblk][n][kpair] so
//   B staging is a straight contiguous copy (async-friendly, no transpose).
// - V pre-transposed to [b,h,d,S] so P@V B-fragments are contiguous global
//   loads (Vt = 24 MB, L2-resident).
// Workspace (~149 MB):
//   xh[16384x768] bf16 | wpa[24][2304][16] u32 | wpp[24][768][16] u32
//   qkvh[16384x2304] bf16 | attnh[16384x768] bf16 | vt[192][64][1024] bf16
// ---------------------------------------------------------------------------

typedef __attribute__((ext_vector_type(16))) __bf16 v16bf;
typedef __attribute__((ext_vector_type(8)))  float  v8f;

union Frag { v16bf v; unsigned u[8]; };

static __device__ inline unsigned short f2bf(float f) {
    union { float f; unsigned u; } x; x.f = f;
    unsigned r = x.u + 0x7FFFu + ((x.u >> 16) & 1u);   // round-to-nearest-even
    return (unsigned short)(r >> 16);
}
static __device__ inline float b2f(unsigned short h) {
    union { unsigned u; float f; } x; x.u = ((unsigned)h) << 16; return x.f;
}
static __device__ inline unsigned pack2(float lo, float hi) {
    return (unsigned)f2bf(lo) | ((unsigned)f2bf(hi) << 16);
}
// A-matrix (16x32 bf16) per-lane K index for packed-pair j (contiguous order):
// lanes 0-15: K = {0..15}; lanes 16-31: +16.
static __device__ inline int a_kpat(int j, int hf) {
    return (j << 1) + (hf << 4);
}

// ---- CDNA5 async global->LDS copy (ASYNCcnt) ------------------------------
#if defined(__has_builtin)
#if __has_builtin(__builtin_amdgcn_global_load_async_to_lds_b128)
#define HAVE_ASYNC_LDS 1
#endif
#endif

#ifdef HAVE_ASYNC_LDS
// builtin expects: (vector_size(16) int AS1*, vector_size(16) int AS3*, Ii, Ii)
typedef int v4i_t __attribute__((__vector_size__(4 * sizeof(int))));
typedef __attribute__((address_space(1))) v4i_t* gas_v4i;
typedef __attribute__((address_space(3))) v4i_t* las_v4i;
static __device__ inline void async_b128(const void* g, void* l) {
    __builtin_amdgcn_global_load_async_to_lds_b128((gas_v4i)g, (las_v4i)l, 0, 0);
}
static __device__ inline void wait_async0() {
#if __has_builtin(__builtin_amdgcn_s_wait_asynccnt)
    __builtin_amdgcn_s_wait_asynccnt(0);
#else
    asm volatile("s_wait_asynccnt 0" ::: "memory");
#endif
}
#endif

// ---------------------------------------------------------------------------
// fp32 -> bf16 conversion (activations)
// ---------------------------------------------------------------------------
__global__ __launch_bounds__(256)
void cvt_f32_bf16(const float* __restrict__ src, unsigned short* __restrict__ dst, int n) {
    int i = blockIdx.x * 256 + threadIdx.x;
    if (i < n) dst[i] = f2bf(src[i]);
}

// ---------------------------------------------------------------------------
// Weight repack: W[K][N] fp32 -> wp[kb][n][kk] u32 (bf16 pair K=2kk,2kk+1).
// This is exactly the LDS layout the GEMM's B-fragment gather wants, so the
// per-tile B copy is contiguous (async b128-able).
// ---------------------------------------------------------------------------
__global__ __launch_bounds__(256)
void pack_w(const float* __restrict__ W, unsigned* __restrict__ wp, int K, int N) {
    int idx = blockIdx.x * 256 + threadIdx.x;
    int total = (K / 32) * N * 16;
    if (idx >= total) return;
    int kk = idx & 15;
    int n  = (idx >> 4) % N;
    int kb = idx / (16 * N);
    int k  = kb * 32 + 2 * kk;
    wp[idx] = pack2(W[(size_t)k * N + n], W[(size_t)(k + 1) * N + n]);
}

// ---------------------------------------------------------------------------
// V transpose: qkv V-section -> vt[(b*12+h)*64 + d][s] bf16 (tiled via LDS)
// ---------------------------------------------------------------------------
__global__ __launch_bounds__(256)
void transpose_v(const unsigned short* __restrict__ qkvh, unsigned short* __restrict__ vt) {
    __shared__ alignas(16) unsigned short tile[256 * 64];   // 32 KB
    const int bh = blockIdx.x;                              // 0..191
    const int b = bh / 12, h = bh % 12;
    const int tid = threadIdx.x;
    for (int s0 = 0; s0 < 1024; s0 += 256) {
        const unsigned* s32 =
            (const unsigned*)(qkvh + (size_t)(b * 1024 + s0 + tid) * 2304 + 1536 + h * 64);
        unsigned* t32 = (unsigned*)&tile[tid * 64];
        #pragma unroll
        for (int i = 0; i < 32; ++i) t32[i] = s32[i];       // 128B/thread contiguous
        __syncthreads();
        for (int i = tid; i < 64 * 128; i += 256) {         // coalesced u32 writes
            int d = i >> 7, sp = i & 127;
            unsigned lo = tile[(2 * sp) * 64 + d];
            unsigned hi = tile[(2 * sp + 1) * 64 + d];
            *(unsigned*)&vt[(size_t)(bh * 64 + d) * 1024 + s0 + 2 * sp] = lo | (hi << 16);
        }
        __syncthreads();
    }
}

// ---------------------------------------------------------------------------
// C[MxN] = A[MxK](bf16 row-major) * Wp(pre-packed) + bias ; 64x64 block,
// 4 waves; per wave: 16-row strip x 4 WMMA n-tiles; K-step 32.
// ---------------------------------------------------------------------------
template <bool OUT_BF16>
__global__ __launch_bounds__(128)
void gemm_bf16_kernel(const unsigned short* __restrict__ A,
                      const unsigned* __restrict__ wp,
                      const float* __restrict__ bias,
                      float* __restrict__ outF,
                      unsigned short* __restrict__ outH,
                      int M, int N, int K)
{
    __shared__ alignas(16) unsigned aS [64 * 16];   // [row][kpair]
    __shared__ alignas(16) unsigned bTs[64 * 16];   // [n][kpair]

    const int tid  = threadIdx.x;
    const int lane = tid & 31;
    const int wave = tid >> 5;
    const int hf   = (lane >> 4) & 1;
    const int ln   = lane & 15;
    const int m0   = blockIdx.y * 64;
    const int n0   = blockIdx.x * 64;

    const v8f zero8 = {0.f,0.f,0.f,0.f,0.f,0.f,0.f,0.f};
    v8f acc[4] = {zero8, zero8, zero8, zero8};

    for (int k0 = 0; k0 < K; k0 += 32) {
        const unsigned* wpblk = wp + ((size_t)(k0 >> 5) * N + n0) * 16; // 4 KB contiguous
#ifdef HAVE_ASYNC_LDS
        // A tile: 64 rows x 64B; chunk c covers LDS bytes [c*16, c*16+16)
        #pragma unroll
        for (int t = 0; t < 2; ++t) {
            int c = tid + t * 128;                 // 0..255
            int r = c >> 2, q = c & 3;
            async_b128(A + (size_t)(m0 + r) * K + k0 + q * 8, (char*)aS + c * 16);
            async_b128((const char*)wpblk + c * 16, (char*)bTs + c * 16);
        }
        wait_async0();
#else
        for (int i = tid; i < 64 * 16; i += 128) {
            int r = i >> 4, c = i & 15;
            aS[i]  = *(const unsigned*)(A + (size_t)(m0 + r) * K + k0 + 2 * c);
            bTs[i] = wpblk[i];
        }
#endif
        __syncthreads();

        Frag a;
        const int ml = wave * 16 + ln;
        #pragma unroll
        for (int j = 0; j < 8; ++j)
            a.u[j] = aS[ml * 16 + (a_kpat(j, hf) >> 1)];

        #pragma unroll
        for (int nt = 0; nt < 4; ++nt) {
            Frag b;
            #pragma unroll
            for (int j = 0; j < 8; ++j)                  // B: K = hf*16 + 2j
                b.u[j] = bTs[(nt * 16 + ln) * 16 + hf * 8 + j];
            acc[nt] = __builtin_amdgcn_wmma_f32_16x16x32_bf16(
                false, a.v, false, b.v, (short)0, acc[nt], false, false);
        }
        __syncthreads();
    }

    // epilogue: C/D layout -> VGPR r is row (r + hf*8), col = ln
    #pragma unroll
    for (int nt = 0; nt < 4; ++nt) {
        int col = n0 + nt * 16 + ln;
        float bv = bias[col];
        #pragma unroll
        for (int r = 0; r < 8; ++r) {
            int row = m0 + wave * 16 + r + hf * 8;
            float v = acc[nt][r] + bv;
            if (OUT_BF16) outH[(size_t)row * N + col] = f2bf(v);
            else          outF[(size_t)row * N + col] = v;
        }
    }
}

// ---------------------------------------------------------------------------
// Causal attention core: one wave per (batch, head, 16-row q-tile).
// qkv rows: [Q(0:768) | K(768:1536) | V(1536:2304)] bf16, row stride 2304.
// vt: V transposed [bh*64+d][S] bf16.
// ---------------------------------------------------------------------------
__global__ __launch_bounds__(32)
void attn_kernel(const unsigned short* __restrict__ qkv,
                 const unsigned short* __restrict__ vt,
                 unsigned short* __restrict__ attnh)
{
    __shared__ alignas(16) unsigned short sc16[16 * 1024]; // P (bf16), 32 KB
    __shared__ float linv[16];                             // per-row 1/sum

    const int lane = threadIdx.x;
    const int hf   = lane >> 4;
    const int ln   = lane & 15;
    const int qt = blockIdx.x & 63;            // q-tile (S/16 = 64)
    const int h  = (blockIdx.x >> 6) % 12;
    const int b  = blockIdx.x / (64 * 12);
    const int s0 = qt * 16;
    const size_t RS = 2304;
    const size_t bbase = (size_t)b * 1024 * RS;

    // Q as two A-fragments (head-dim K = 64 = 2 x 32), pairs = aligned u32
    Frag qf[2];
    {
        const unsigned short* qrow = qkv + bbase + (size_t)(s0 + ln) * RS + h * 64;
        #pragma unroll
        for (int f = 0; f < 2; ++f)
            #pragma unroll
            for (int j = 0; j < 8; ++j)
                qf[f].u[j] = *(const unsigned*)(qrow + f * 32 + a_kpat(j, hf));
    }

    const v8f zero8 = {0.f,0.f,0.f,0.f,0.f,0.f,0.f,0.f};

    // ---- scores = (Q K^T) / 8, causal-masked, stored bf16 in LDS ----
    for (int kt = 0; kt <= qt; ++kt) {
        const unsigned short* krow =
            qkv + bbase + (size_t)(kt * 16 + ln) * RS + 768 + h * 64;
        v8f c = zero8;
        #pragma unroll
        for (int f = 0; f < 2; ++f) {
            Frag bf_;                       // B[k][n] = K[n][k] -> read K row n
            #pragma unroll
            for (int j = 0; j < 8; ++j)
                bf_.u[j] = *(const unsigned*)(krow + f * 32 + hf * 16 + 2 * j);
            c = __builtin_amdgcn_wmma_f32_16x16x32_bf16(
                false, qf[f].v, false, bf_.v, (short)0, c, false, false);
        }
        #pragma unroll
        for (int r = 0; r < 8; ++r) {
            int m = r + hf * 8;
            int kp = kt * 16 + ln;
            float v = c[r] * 0.125f;                 // 1/sqrt(64)
            if (kp > s0 + m) v = -1e30f;             // causal mask
            sc16[m * 1024 + kp] = f2bf(v);
        }
    }
    const int L   = (qt + 1) * 16;
    const int L32 = (L + 31) & ~31;
    if (L32 > L)                                     // zero-pad to k-step
        for (int i = lane; i < 16 * 16; i += 32)
            sc16[(i >> 4) * 1024 + L + (i & 15)] = 0;
    __syncthreads();

    // ---- softmax (fp32 math, bf16 storage); 1/sum applied at epilogue ----
    for (int m = 0; m < 16; ++m) {
        float mx = -1e30f;
        for (int c = lane; c < L; c += 32) mx = fmaxf(mx, b2f(sc16[m * 1024 + c]));
        #pragma unroll
        for (int off = 16; off >= 1; off >>= 1) mx = fmaxf(mx, __shfl_xor(mx, off, 32));
        float sum = 0.f;
        for (int c = lane; c < L; c += 32) {
            float e = __expf(b2f(sc16[m * 1024 + c]) - mx);
            sc16[m * 1024 + c] = f2bf(e);
            sum += e;
        }
        #pragma unroll
        for (int off = 16; off >= 1; off >>= 1) sum += __shfl_xor(sum, off, 32);
        if (lane == 0) linv[m] = 1.f / sum;
    }
    __syncthreads();

    // ---- out = P V : A = P rows from LDS, B = contiguous Vt rows ----
    const unsigned short* vbase = vt + (size_t)(b * 12 + h) * 64 * 1024;
    v8f o[4] = {zero8, zero8, zero8, zero8};
    for (int k0 = 0; k0 < L32; k0 += 32) {
        Frag a;
        #pragma unroll
        for (int j = 0; j < 8; ++j)
            a.u[j] = *(const unsigned*)&sc16[ln * 1024 + k0 + a_kpat(j, hf)];
        #pragma unroll
        for (int nt = 0; nt < 4; ++nt) {
            const unsigned short* vrow =
                vbase + (size_t)(nt * 16 + ln) * 1024 + k0 + hf * 16;
            Frag bv;
            #pragma unroll
            for (int j = 0; j < 8; ++j)
                bv.u[j] = *(const unsigned*)(vrow + 2 * j);   // 2x b128 / frag
            o[nt] = __builtin_amdgcn_wmma_f32_16x16x32_bf16(
                false, a.v, false, bv.v, (short)0, o[nt], false, false);
        }
    }

    // ---- epilogue: normalize rows, write bf16 attn activations ----
    #pragma unroll
    for (int nt = 0; nt < 4; ++nt)
        #pragma unroll
        for (int r = 0; r < 8; ++r) {
            int m = r + hf * 8;
            float v = o[nt][r] * linv[m];
            size_t row = (size_t)(b * 1024 + s0 + m);
            attnh[row * 768 + h * 64 + nt * 16 + ln] = f2bf(v);
        }
}

// ---------------------------------------------------------------------------
extern "C" void kernel_launch(void* const* d_in, const int* in_sizes, int n_in,
                              void* d_out, int out_size, void* d_ws, size_t ws_size,
                              hipStream_t stream) {
    (void)in_sizes; (void)n_in; (void)out_size; (void)ws_size;
    const float* x      = (const float*)d_in[0];
    const float* W_attn = (const float*)d_in[1];
    const float* b_attn = (const float*)d_in[2];
    const float* W_proj = (const float*)d_in[3];
    const float* b_proj = (const float*)d_in[4];
    float* out = (float*)d_out;

    const int M = 16 * 1024;          // 16384 tokens
    char* ws = (char*)d_ws;
    unsigned short* xh    = (unsigned short*)ws; ws += (size_t)M * 768 * 2;
    unsigned*       wpa   = (unsigned*)ws;       ws += (size_t)24 * 2304 * 16 * 4;
    unsigned*       wpp   = (unsigned*)ws;       ws += (size_t)24 * 768 * 16 * 4;
    unsigned short* qkvh  = (unsigned short*)ws; ws += (size_t)M * 2304 * 2;
    unsigned short* attnh = (unsigned short*)ws; ws += (size_t)M * 768 * 2;
    unsigned short* vt    = (unsigned short*)ws; ws += (size_t)192 * 64 * 1024 * 2;

    int nx = M * 768;
    int npa = 24 * 2304 * 16, npp = 24 * 768 * 16;
    cvt_f32_bf16<<<(nx + 255) / 256, 256, 0, stream>>>(x, xh, nx);
    pack_w<<<(npa + 255) / 256, 256, 0, stream>>>(W_attn, wpa, 768, 2304);
    pack_w<<<(npp + 255) / 256, 256, 0, stream>>>(W_proj, wpp, 768, 768);

    gemm_bf16_kernel<true><<<dim3(2304 / 64, M / 64), 128, 0, stream>>>(
        xh, wpa, b_attn, nullptr, qkvh, M, 2304, 768);

    transpose_v<<<192, 256, 0, stream>>>(qkvh, vt);

    attn_kernel<<<16 * 12 * 64, 32, 0, stream>>>(qkvh, vt, attnh);

    gemm_bf16_kernel<false><<<dim3(768 / 64, M / 64), 128, 0, stream>>>(
        attnh, wpp, b_proj, out, nullptr, M, 768, 768);
}
